// MSFModule_53171695124819
// MI455X (gfx1250) — compile-verified, matrix operand order checked by reading, not run on hardware
//
#include <hip/hip_runtime.h>
#include <hip/hip_bf16.h>
#include <stddef.h>

// ---------------------------------------------------------------------------
// Self-attention (SAGAN style) fused implementation for gfx1250 (MI455X).
// All GEMMs run on v_wmma_f32_16x16x32_f16 (f16 in, f32 accumulate).
// Attention is flash-style: energy (B,N,N) is never materialized in HBM.
// K/V panels are staged into LDS once per workgroup per key-step via
// GLOBAL_LOAD_ASYNC_TO_LDS_B128 (ASYNCcnt) when the builtin is available.
// ---------------------------------------------------------------------------

typedef _Float16 half_t;
typedef __attribute__((ext_vector_type(16))) _Float16 v16h;
typedef __attribute__((ext_vector_type(8)))  float    v8f;
typedef __attribute__((ext_vector_type(4)))  unsigned int u32x4;
typedef int v4i_gcc __attribute__((vector_size(16)));   // builtin param type

static constexpr int Bb = 4;
static constexpr int Cc = 256;
static constexpr int Nn = 4096;   // 64*64
static constexpr int C8 = 32;

union H16 { v16h h; u32x4 u[2]; };

// Load 16 halves in the 16-bit A/B WMMA operand layout:
//   VGPR0-3 <- K = base..base+7   (16 contiguous bytes)
//   VGPR4-7 <- K = base+16..base+23
// Caller passes p already offset by (row*stride + coff) where
// coff = (lane<16 ? 0 : 8).
__device__ __forceinline__ v16h ld16(const half_t* p) {
    H16 r;
    r.u[0] = *(const u32x4*)(p);
    r.u[1] = *(const u32x4*)(p + 16);
    return r.h;
}

__device__ __forceinline__ v8f wmma32(v16h a, v16h b, v8f c) {
    return __builtin_amdgcn_wmma_f32_16x16x32_f16(
        /*neg_a=*/false, a, /*neg_b=*/false, b,
        /*c_mod=*/(short)0, c, /*reuse_a=*/false, /*reuse_b=*/false);
}

// 16-byte global -> LDS copy. Uses the gfx1250 async path
// (GLOBAL_LOAD_ASYNC_TO_LDS_B128, tracked by ASYNCcnt) when available,
// else a register bounce (global_load_b128 + ds_store_b128).
__device__ __forceinline__ void cp16_g2l(const half_t* g, half_t* l) {
#if defined(__HIP_DEVICE_COMPILE__) && __has_builtin(__builtin_amdgcn_global_load_async_to_lds_b128)
    __builtin_amdgcn_global_load_async_to_lds_b128(
        (__attribute__((address_space(1))) v4i_gcc*)g,
        (__attribute__((address_space(3))) v4i_gcc*)l,
        /*offset=*/0, /*cpol=*/0);
#else
    *(u32x4*)l = *(const u32x4*)g;
#endif
}

__device__ __forceinline__ void wait_g2l() {
#if defined(__HIP_DEVICE_COMPILE__) && __has_builtin(__builtin_amdgcn_global_load_async_to_lds_b128)
#if __has_builtin(__builtin_amdgcn_s_wait_asynccnt)
    __builtin_amdgcn_s_wait_asynccnt(0);
#else
    asm volatile("s_wait_asynccnt 0" ::: "memory");
#endif
#endif
    // register-bounce fallback is ordered by the compiler's dscnt/loadcnt
    // bookkeeping + the following __syncthreads()
}

// ---------------------------------------------------------------------------
// Kernel 0: fp32 -> f16 conversions.
//   xth[b][n][c] = (f16) x[b][c][n]   (B,N,C) layout: contiguous reduction dim
//   Wqh/Wkh (32x256), Wvh/Woh (256x256) row-major f16
// ---------------------------------------------------------------------------
__global__ __launch_bounds__(256) void convert_kernel(
    const float* __restrict__ x,
    const float* __restrict__ Wq, const float* __restrict__ Wk,
    const float* __restrict__ Wv, const float* __restrict__ Wo,
    half_t* __restrict__ xth,
    half_t* __restrict__ Wqh, half_t* __restrict__ Wkh,
    half_t* __restrict__ Wvh, half_t* __restrict__ Woh)
{
    const size_t tid = (size_t)blockIdx.x * blockDim.x + threadIdx.x;
    const size_t xelems = (size_t)Bb * Cc * Nn;
    if (tid < xelems) {
        // coalesced read over n, scattered f16 write (transpose)
        const size_t b = tid / ((size_t)Cc * Nn);
        const size_t r = tid % ((size_t)Cc * Nn);
        const size_t c = r / Nn;
        const size_t n = r % Nn;
        xth[(b * Nn + n) * Cc + c] = (half_t)x[tid];
    }
    if (tid < (size_t)C8 * Cc) {
        Wqh[tid] = (half_t)Wq[tid];
        Wkh[tid] = (half_t)Wk[tid];
    }
    if (tid < (size_t)Cc * Cc) {
        Wvh[tid] = (half_t)Wv[tid];
        Woh[tid] = (half_t)Wo[tid];
    }
}

// ---------------------------------------------------------------------------
// Kernel 1: q/k/v projections as WMMA GEMMs.
// One wave (32 threads) computes a 16(n) x 32(outdim) tile.
// Jobs per 16-row block: 0 -> q (Wq), 1 -> k (Wk), 2..9 -> v chunk (Wv).
//   qh, kh stored (B, N, 32)          (contiguous d; K panels of 32 rows are
//                                      also 2KB-contiguous for LDS staging)
//   vh     stored (B, N/32, C, 32)    (each 32-key V panel = contiguous 16KB)
// ---------------------------------------------------------------------------
__global__ __launch_bounds__(32) void qkv_kernel(
    const half_t* __restrict__ xth,
    const half_t* __restrict__ Wqh, const half_t* __restrict__ Wkh,
    const half_t* __restrict__ Wvh,
    const float* __restrict__ bq, const float* __restrict__ bk,
    const float* __restrict__ bv,
    half_t* __restrict__ qh, half_t* __restrict__ kh, half_t* __restrict__ vh)
{
    const int lane = threadIdx.x & 31;
    const int l15  = lane & 15;
    const int hi   = lane >> 4;
    const int coff = hi ? 8 : 0;

    const int job = blockIdx.x;
    const int jt  = job % 10;
    const int nb  = (job / 10) % (Nn / 16);
    const int b   = job / (10 * (Nn / 16));
    const int nbase = nb * 16;

    const half_t* A = xth + ((size_t)b * Nn + nbase) * Cc;

    const half_t* W;
    const float*  bias;
    int colbase;
    if (jt == 0)      { W = Wqh; bias = bq; colbase = 0; }
    else if (jt == 1) { W = Wkh; bias = bk; colbase = 0; }
    else              { W = Wvh; bias = bv; colbase = (jt - 2) * 32; }

    v8f c0 = {}, c1 = {};
    #pragma unroll
    for (int cc = 0; cc < Cc; cc += 32) {
        v16h a  = ld16(A + (size_t)l15 * Cc + cc + coff);
        v16h b0 = ld16(W + (size_t)(colbase + l15)      * Cc + cc + coff);
        v16h b1 = ld16(W + (size_t)(colbase + 16 + l15) * Cc + cc + coff);
        c0 = wmma32(a, b0, c0);
        c1 = wmma32(a, b1, c1);
    }

    const float bias0 = bias[colbase + l15];
    const float bias1 = bias[colbase + 16 + l15];

    if (jt < 2) {
        half_t* out = (jt == 0 ? qh : kh) + ((size_t)b * Nn + nbase) * C8;
        #pragma unroll
        for (int g = 0; g < 8; ++g) {
            const int row = g + 8 * hi;
            out[(size_t)row * C8 + l15]      = (half_t)(c0[g] + bias0);
            out[(size_t)row * C8 + 16 + l15] = (half_t)(c1[g] + bias1);
        }
    } else {
        // blocked V: vh[b][n/32][c][n%32]
        half_t* outv = vh + (size_t)b * (Nn / 32) * Cc * 32;
        #pragma unroll
        for (int g = 0; g < 8; ++g) {
            const int n   = nbase + g + 8 * hi;
            const int nb2 = n >> 5;
            const int jl  = n & 31;
            outv[((size_t)nb2 * Cc + colbase + l15)      * 32 + jl] = (half_t)(c0[g] + bias0);
            outv[((size_t)nb2 * Cc + colbase + 16 + l15) * 32 + jl] = (half_t)(c1[g] + bias1);
        }
    }
}

// ---------------------------------------------------------------------------
// Kernel 2: flash attention.
// Block = (b, 32-query block). 256 threads = 8 waves:
//   wave = strip(2: 16 query rows each) x channel-group(4: 64 channels each)
// Per 32-key step:
//   - stage K (2KB) + V (16KB) panels into LDS once for the whole workgroup
//     (async-to-LDS), serving all 8 waves via ds_load_b128
//   - 2 WMMAs (S = Q K), online softmax via 16-lane shuffle reductions,
//     P -> per-wave LDS -> A-layout, 4 WMMAs (O += P V^T).
// O written as (B, N, C) f16.
// ---------------------------------------------------------------------------
__global__ __launch_bounds__(256) void attn_kernel(
    const half_t* __restrict__ qh, const half_t* __restrict__ kh,
    const half_t* __restrict__ vh, half_t* __restrict__ oh)
{
    const int t    = threadIdx.x;
    const int wave = t >> 5;
    const int lane = t & 31;
    const int l15  = lane & 15;
    const int hi   = lane >> 4;
    const int coff = hi ? 8 : 0;

    const int strip = wave >> 2;      // 0..1
    const int cg    = wave & 3;       // 0..3
    const int qblk  = blockIdx.x % (Nn / 32);
    const int b     = blockIdx.x / (Nn / 32);
    const int qbase = qblk * 32 + strip * 16;
    const int cbase = cg * 64;

    const half_t* qb  = qh + (size_t)b * Nn * C8;
    const half_t* kb  = kh + (size_t)b * Nn * C8;
    const half_t* vbB = vh + (size_t)b * (Nn / 32) * Cc * 32;

    // Q strip as WMMA A operand (held in registers for the whole loop)
    const v16h qa = ld16(qb + (size_t)(qbase + l15) * C8 + coff);

    __shared__ half_t kTile[32 * 32];        // K panel: [j_local][d]    2KB
    __shared__ half_t vTile[Cc * 32];        // V panel: [c][j_local]   16KB
    __shared__ half_t plds[8 * 16 * 32];     // per-wave 16x32 f16 P tile 8KB
    half_t* pw = plds + wave * (16 * 32);

    float mrun[8], lrun[8];
    v8f o0 = {}, o1 = {}, o2 = {}, o3 = {};
    #pragma unroll
    for (int g = 0; g < 8; ++g) { mrun[g] = -1e30f; lrun[g] = 0.0f; }

    for (int j0 = 0; j0 < Nn; j0 += 32) {
        __syncthreads();   // previous step's LDS reads complete

        // ---- stage K panel: 1024 halves = 128 x 16B chunks (contiguous)
        if (t < 128)
            cp16_g2l(kb + (size_t)j0 * C8 + t * 8, kTile + t * 8);
        // ---- stage V panel: 8192 halves = 1024 x 16B chunks (contiguous)
        {
            const half_t* vblk = vbB + (size_t)(j0 >> 5) * Cc * 32;
            #pragma unroll
            for (int k2 = 0; k2 < 4; ++k2)
                cp16_g2l(vblk + (size_t)(t + k2 * 256) * 8,
                         vTile + (t + k2 * 256) * 8);
        }
        wait_g2l();
        __syncthreads();   // panels visible to all waves

        // ---- S = Q K : two 16x16 f32 tiles (cols j0..15, j0+16..31)
        v16h kt0 = ld16(kTile + l15 * 32 + coff);
        v16h kt1 = ld16(kTile + (16 + l15) * 32 + coff);
        v8f s0 = {}, s1 = {};
        s0 = wmma32(qa, kt0, s0);
        s1 = wmma32(qa, kt1, s1);

        // ---- online softmax (rows live in 16-lane halves, slot g = row g+8*hi)
        float alpha[8];
        #pragma unroll
        for (int g = 0; g < 8; ++g) {
            float m = fmaxf(s0[g], s1[g]);
            m = fmaxf(m, __shfl_xor(m, 1));
            m = fmaxf(m, __shfl_xor(m, 2));
            m = fmaxf(m, __shfl_xor(m, 4));
            m = fmaxf(m, __shfl_xor(m, 8));
            const float mnew = fmaxf(mrun[g], m);
            alpha[g] = __expf(mrun[g] - mnew);
            mrun[g]  = mnew;
        }
        #pragma unroll
        for (int g = 0; g < 8; ++g) {
            const float p0 = __expf(s0[g] - mrun[g]);
            const float p1 = __expf(s1[g] - mrun[g]);
            const int row = g + 8 * hi;
            pw[row * 32 + l15]      = (half_t)p0;   // C-layout -> LDS
            pw[row * 32 + 16 + l15] = (half_t)p1;
            float r = p0 + p1;
            r += __shfl_xor(r, 1);
            r += __shfl_xor(r, 2);
            r += __shfl_xor(r, 4);
            r += __shfl_xor(r, 8);
            lrun[g] = lrun[g] * alpha[g] + r;
        }

        // ---- rescale running O by alpha
        #pragma unroll
        for (int g = 0; g < 8; ++g) {
            o0[g] *= alpha[g]; o1[g] *= alpha[g];
            o2[g] *= alpha[g]; o3[g] *= alpha[g];
        }

        // ---- P back from LDS in A-operand layout (DS ops in-order per wave)
        asm volatile("s_wait_dscnt 0" ::: "memory");
        v16h pa = ld16(pw + l15 * 32 + coff);

        // ---- O += P V^T : four 16x16 channel tiles, V from LDS
        v16h vt0 = ld16(vTile + (size_t)(cbase +  0 + l15) * 32 + coff);
        v16h vt1 = ld16(vTile + (size_t)(cbase + 16 + l15) * 32 + coff);
        v16h vt2 = ld16(vTile + (size_t)(cbase + 32 + l15) * 32 + coff);
        v16h vt3 = ld16(vTile + (size_t)(cbase + 48 + l15) * 32 + coff);
        o0 = wmma32(pa, vt0, o0);
        o1 = wmma32(pa, vt1, o1);
        o2 = wmma32(pa, vt2, o2);
        o3 = wmma32(pa, vt3, o3);
    }

    // ---- normalize and store O as (B, N, C) f16
    half_t* ob = oh + (size_t)b * Nn * Cc;
    #pragma unroll
    for (int g = 0; g < 8; ++g) {
        const float inv = __frcp_rn(lrun[g]);
        const int row = qbase + g + 8 * hi;
        ob[(size_t)row * Cc + cbase +  0 + l15] = (half_t)(o0[g] * inv);
        ob[(size_t)row * Cc + cbase + 16 + l15] = (half_t)(o1[g] * inv);
        ob[(size_t)row * Cc + cbase + 32 + l15] = (half_t)(o2[g] * inv);
        ob[(size_t)row * Cc + cbase + 48 + l15] = (half_t)(o3[g] * inv);
    }
}

// ---------------------------------------------------------------------------
// Kernel 3: residual + output projection.
//   out[b,d,n] = sum_c Wo[d,c] * (O[b,n,c] + xT[b,n,c]) + bo[d]
// One wave per 16(n) x 32(d) tile; A = packed-f16 add of O and xT tiles.
// ---------------------------------------------------------------------------
__global__ __launch_bounds__(32) void outproj_kernel(
    const half_t* __restrict__ oh, const half_t* __restrict__ xth,
    const half_t* __restrict__ Woh, const float* __restrict__ bo,
    float* __restrict__ out)
{
    const int lane = threadIdx.x & 31;
    const int l15  = lane & 15;
    const int hi   = lane >> 4;
    const int coff = hi ? 8 : 0;

    const int job = blockIdx.x;
    const int dc  = job % (Cc / 32);
    const int nb  = (job / (Cc / 32)) % (Nn / 16);
    const int b   = job / ((Cc / 32) * (Nn / 16));
    const int nbase   = nb * 16;
    const int colbase = dc * 32;

    const half_t* O = oh  + ((size_t)b * Nn + nbase) * Cc;
    const half_t* X = xth + ((size_t)b * Nn + nbase) * Cc;

    v8f c0 = {}, c1 = {};
    #pragma unroll
    for (int cc = 0; cc < Cc; cc += 32) {
        v16h a1 = ld16(O + (size_t)l15 * Cc + cc + coff);
        v16h a2 = ld16(X + (size_t)l15 * Cc + cc + coff);
        v16h a  = a1 + a2;                       // v_pk_add_f16 (residual)
        v16h b0 = ld16(Woh + (size_t)(colbase + l15)      * Cc + cc + coff);
        v16h b1 = ld16(Woh + (size_t)(colbase + 16 + l15) * Cc + cc + coff);
        c0 = wmma32(a, b0, c0);
        c1 = wmma32(a, b1, c1);
    }

    const float bias0 = bo[colbase + l15];
    const float bias1 = bo[colbase + 16 + l15];
    float* ob = out + (size_t)b * Cc * Nn;
    #pragma unroll
    for (int g = 0; g < 8; ++g) {
        const int n = nbase + g + 8 * hi;
        ob[(size_t)(colbase + l15)      * Nn + n] = c0[g] + bias0;
        ob[(size_t)(colbase + 16 + l15) * Nn + n] = c1[g] + bias1;
    }
}

// ---------------------------------------------------------------------------
// Host-side launcher
// ---------------------------------------------------------------------------
extern "C" void kernel_launch(void* const* d_in, const int* in_sizes, int n_in,
                              void* d_out, int out_size, void* d_ws, size_t ws_size,
                              hipStream_t stream)
{
    (void)in_sizes; (void)n_in; (void)out_size; (void)ws_size;

    const float* x  = (const float*)d_in[0];
    const float* Wq = (const float*)d_in[1];
    const float* bq = (const float*)d_in[2];
    const float* Wk = (const float*)d_in[3];
    const float* bk = (const float*)d_in[4];
    const float* Wv = (const float*)d_in[5];
    const float* bv = (const float*)d_in[6];
    const float* Wo = (const float*)d_in[7];
    const float* bo = (const float*)d_in[8];
    float* out = (float*)d_out;

    // workspace layout (f16), ~26 MB total
    half_t* xth = (half_t*)d_ws;                          // B*N*C
    half_t* oh  = xth + (size_t)Bb * Nn * Cc;             // B*N*C
    half_t* qh  = oh  + (size_t)Bb * Nn * Cc;             // B*N*32
    half_t* kh  = qh  + (size_t)Bb * Nn * C8;             // B*N*32
    half_t* vh  = kh  + (size_t)Bb * Nn * C8;             // B*(N/32)*C*32
    half_t* Wqh = vh  + (size_t)Bb * Cc * Nn;             // 32*256
    half_t* Wkh = Wqh + (size_t)C8 * Cc;
    half_t* Wvh = Wkh + (size_t)C8 * Cc;                  // 256*256
    half_t* Woh = Wvh + (size_t)Cc * Cc;

    // 0) convert to f16 (x transposed to (B,N,C))
    {
        const size_t total = (size_t)Bb * Cc * Nn;
        const int blocks = (int)((total + 255) / 256);
        convert_kernel<<<blocks, 256, 0, stream>>>(
            x, Wq, Wk, Wv, Wo, xth, Wqh, Wkh, Wvh, Woh);
    }

    // 1) q/k/v projections (WMMA)
    {
        const int blocks = Bb * (Nn / 16) * 10;           // 10240 waves
        qkv_kernel<<<blocks, 32, 0, stream>>>(
            xth, Wqh, Wkh, Wvh, bq, bk, bv, qh, kh, vh);
    }

    // 2) fused flash attention (WMMA + async-LDS-staged K/V panels)
    {
        const int blocks = Bb * (Nn / 32);                // 512 blocks x 8 waves
        attn_kernel<<<blocks, 256, 0, stream>>>(qh, kh, vh, oh);
    }

    // 3) residual + output projection (WMMA)
    {
        const int blocks = Bb * (Nn / 16) * (Cc / 32);    // 8192 waves
        outproj_kernel<<<blocks, 32, 0, stream>>>(oh, xth, Woh, bo, out);
    }
}
